// HGTModel_77902116815126
// MI455X (gfx1250) — matrix operand order repaired and checked by reading, not compiled, not run on hardware
//
#include <hip/hip_runtime.h>
#include <hip/hip_bf16.h>
#include <climits>
#include <cstdint>

#define HID 128
#define NHEAD 8
#define DHEAD 16

#define BM 128
#define BN 64
#define BK 32

typedef __attribute__((ext_vector_type(16))) __bf16 v16bf;
typedef __attribute__((ext_vector_type(8)))  float  v8f;
typedef __attribute__((ext_vector_type(4)))  unsigned int v4u;
typedef __attribute__((ext_vector_type(8)))  int v8i;
typedef __attribute__((ext_vector_type(4)))  int v4i;
typedef unsigned int u32;

// Order-preserving float -> int key for atomicMax-based segment max.
__device__ __forceinline__ int fkey_encode(float f) {
    int i = __float_as_int(f);
    return (i >= 0) ? i : (i ^ 0x7FFFFFFF);
}
__device__ __forceinline__ float fkey_decode(int k) {
    return __int_as_float((k >= 0) ? k : (k ^ 0x7FFFFFFF));
}

// pack two floats into a dword of bf16s (RNE) -> v_cvt_pk_bf16_f32
__device__ __forceinline__ u32 pack_bf16(float lo, float hi) {
    __bf16 l = (__bf16)lo, h = (__bf16)hi;
    unsigned short ul = __builtin_bit_cast(unsigned short, l);
    unsigned short uh = __builtin_bit_cast(unsigned short, h);
    return (u32)ul | ((u32)uh << 16);
}

// ---------------------------------------------------------------------------
// TDM: async 2D tile load Global -> LDS, fp32 elements.
// Descriptor bit layout per CDNA5 ISA 8.3-8.6 (D# groups 0..3).
// OOB rows/cols (beyond tensor_x/tensor_y) are returned as ZERO by the TDM,
// which implements the M-edge zero-fill for the GEMM in hardware.
// ---------------------------------------------------------------------------
__device__ __forceinline__ void tdm_load_2d(u32 lds_off, const void* gaddr,
                                            u32 tile_x, u32 tile_y,
                                            u32 tensor_x, u32 tensor_y,
                                            u32 stride_elems)
{
    unsigned long long ga = (unsigned long long)(uintptr_t)gaddr;
    v4u g0;
    g0[0] = 1u;                                         // count=1, user mode
    g0[1] = lds_off;                                    // LDS byte address
    g0[2] = (u32)ga;                                    // global_addr[31:0]
    g0[3] = (u32)((ga >> 32) & 0x01FFFFFFu) | (2u << 30); // addr[56:32] | type=2
    v8i g1;
    g1[0] = (int)(2u << 16);                            // wg_mask=0, data_size=4B
    g1[1] = (int)((tensor_x & 0xFFFFu) << 16);          // tensor_dim0[15:0]
    g1[2] = (int)((tensor_x >> 16) | ((tensor_y & 0xFFFFu) << 16));
    g1[3] = (int)((tensor_y >> 16) | (tile_x << 16));   // tensor_dim1 hi | tile_dim0
    g1[4] = (int)(tile_y & 0xFFFFu);                    // tile_dim1, tile_dim2=0
    g1[5] = (int)stride_elems;                          // tensor_dim0_stride[31:0]
    g1[6] = 0;                                          // stride0 hi, stride1 lo
    g1[7] = 0;                                          // stride1 hi
    v4i g2 = {1, 1, 0, 0};                              // tensor_dim2=1, dim3=1
    v4i g3 = {0, 0, 0, 0};
#if defined(__clang_major__) && (__clang_major__ >= 23)
    v8i g4 = {0, 0, 0, 0, 0, 0, 0, 0};
    __builtin_amdgcn_tensor_load_to_lds(g0, g1, g2, g3, g4, 0);
#else
    __builtin_amdgcn_tensor_load_to_lds(g0, g1, g2, g3, 0);
#endif
}

// ---------------------------------------------------------------------------
// Tiled WMMA GEMM: C[M,N] = epilogue(A[M,K] @ B[K,N] + bias)
// Block = 256 threads (8 waves) computes a BM x BN = 128x64 tile of C.
// A tile (128x32 fp32) is DMA'd into LDS by the Tensor Data Mover each K-step
// (OOB rows auto-zero); B tile (32x64) is staged transposed as packed bf16.
// Each wave owns 16 rows and reuses its A fragment across 4 WMMA tiles.
// mode 0: none; 1: relu; 2: relu(g*y + (1-g)*Xprev), g = sigmoid(*skip)
// Requires: N % 64 == 0, K % 32 == 0 (holds for this model).
// ---------------------------------------------------------------------------
__global__ void __launch_bounds__(256)
gemm_bf16_wmma(const float* __restrict__ A, int lda,
               const float* __restrict__ B, int ldb,
               const float* __restrict__ bias,
               float* __restrict__ C, int ldc,
               const float* __restrict__ Xprev,
               const float* __restrict__ skip,
               int M, int N, int K, int mode)
{
    __shared__ __align__(16) float lds_a[BM][BK];       // fp32, filled by TDM
    __shared__ __align__(16) u32   lds_b[BN][BK / 2];   // bf16 pairs, [n][kpair]

    const int tid  = threadIdx.x;
    const int wave = tid >> 5;
    const int lane = tid & 31;
    const int hb   = lane >> 4;        // half-wave: 0 or 1
    const int mr   = lane & 15;

    const int blockM = blockIdx.y * BM;
    const int blockN = blockIdx.x * BN;
    const int tileM  = blockM + wave * 16;   // this wave's 16 rows

    v8f zero = {0.f, 0.f, 0.f, 0.f, 0.f, 0.f, 0.f, 0.f};
    v8f acc[4] = {zero, zero, zero, zero};

    const u32 lds_a_off = (u32)(uintptr_t)&lds_a[0][0];

    // B staging coordinates: thread -> (column n, 8-row k slab)
    const int bn = tid & 63;           // B col 0..63
    const int bk = (tid >> 6) * 8;     // B k base: 0,8,16,24

    for (int k0 = 0; k0 < K; k0 += BK) {
        // ---- A tile via Tensor Data Mover (wave 0 issues + waits) ----
        if (tid < 32) {
            tdm_load_2d(lds_a_off, A + (size_t)blockM * lda + k0,
                        /*tile_x=*/BK, /*tile_y=*/BM,
                        /*tensor_x=*/(u32)(lda - k0), /*tensor_y=*/(u32)(M - blockM),
                        /*stride=*/(u32)lda);
            __builtin_amdgcn_s_wait_tensorcnt(0);
        }
        // ---- B tile staged transposed as packed bf16 (one b128 store) ----
        {
            const float* src = B + (size_t)(k0 + bk) * ldb + blockN + bn;
            if (k0 + BK < K) __builtin_prefetch(src + (size_t)BK * ldb, 0, 1);
            float f0 = src[0];
            float f1 = src[(size_t)ldb];
            float f2 = src[(size_t)2 * ldb];
            float f3 = src[(size_t)3 * ldb];
            float f4 = src[(size_t)4 * ldb];
            float f5 = src[(size_t)5 * ldb];
            float f6 = src[(size_t)6 * ldb];
            float f7 = src[(size_t)7 * ldb];
            uint4 pk;
            pk.x = pack_bf16(f0, f1);
            pk.y = pack_bf16(f2, f3);
            pk.z = pack_bf16(f4, f5);
            pk.w = pack_bf16(f6, f7);
            *(uint4*)&lds_b[bn][bk >> 1] = pk;
        }
        __syncthreads();   // publish TDM-written A + staged B

        // ---- A fragment (ISA 7.12.2 16-bit A 16x32 layout), cvt fp32->bf16 ----
        v16bf a;
        {
            const float* arow = &lds_a[wave * 16 + mr][0];
#pragma unroll
            for (int r = 0; r < 8; ++r) {
                const int ka = ((r < 4) ? (2 * r) : (2 * r + 8)) + hb * 8;
                a[2 * r]     = (__bf16)arow[ka];
                a[2 * r + 1] = (__bf16)arow[ka + 1];
            }
        }
        // ---- 4 WMMA tiles along N, reusing the A fragment ----
#pragma unroll
        for (int nt = 0; nt < 4; ++nt) {
            union { u32 u[8]; v16bf v; } bu;
            const u32* brow = &lds_b[nt * 16 + mr][hb * 4];
            uint4 q0 = *(const uint4*)(brow);
            uint4 q1 = *(const uint4*)(brow + 8);
            bu.u[0] = q0.x; bu.u[1] = q0.y; bu.u[2] = q0.z; bu.u[3] = q0.w;
            bu.u[4] = q1.x; bu.u[5] = q1.y; bu.u[6] = q1.z; bu.u[7] = q1.w;
            acc[nt] = __builtin_amdgcn_wmma_f32_16x16x32_bf16(
                false, a, false, bu.v, (short)0, acc[nt], false, false);
        }
        __syncthreads();   // fragment reads done before next staging
    }

    // ---- epilogue: C reg j, half hb -> row M = j + hb*8, col N = lane&15 ----
    float g = 0.0f;
    if (mode == 2) g = 1.0f / (1.0f + expf(-skip[0]));
#pragma unroll
    for (int nt = 0; nt < 4; ++nt) {
        const int n = blockN + nt * 16 + mr;
        const float bv = bias ? bias[n] : 0.0f;
#pragma unroll
        for (int j = 0; j < 8; ++j) {
            const int m = tileM + hb * 8 + j;
            if (m < M) {
                float y = acc[nt][j] + bv;
                if (mode == 1) {
                    y = fmaxf(y, 0.0f);
                } else if (mode == 2) {
                    y = g * y + (1.0f - g) * Xprev[(size_t)m * ldc + n];
                    y = fmaxf(y, 0.0f);
                }
                C[(size_t)m * ldc + n] = y;
            }
        }
    }
}

// ---------------------------------------------------------------------------
// Per-node, per-relation head transform: out[n,h,f] = sum_d src[n,h,d]*A[h,d,f]
// Hoisted out of the per-edge einsum (A depends only on relation, not edge).
// ---------------------------------------------------------------------------
__global__ void head_transform_kernel(const float* __restrict__ src, int src_ld,
                                      int col_ofs,
                                      const float* __restrict__ Amat, // [H,D,D]
                                      float* __restrict__ out, int n_nodes)
{
    int idx = blockIdx.x * blockDim.x + threadIdx.x;
    if (idx >= n_nodes * HID) return;
    int n = idx >> 7;
    int c = idx & 127;
    int h = c >> 4, f = c & 15;
    const float* s   = src + (size_t)n * src_ld + col_ofs + h * DHEAD;
    const float* Ahd = Amat + h * DHEAD * DHEAD + f;
    float acc = 0.0f;
#pragma unroll
    for (int d = 0; d < DHEAD; ++d) acc += s[d] * Ahd[d * DHEAD];
    out[(size_t)n * HID + c] = acc;
}

__global__ void init_agg_kernel(int* __restrict__ mkey, float* __restrict__ den,
                                float* __restrict__ agg, int n_nodes)
{
    int idx = blockIdx.x * blockDim.x + threadIdx.x;
    if (idx < n_nodes * NHEAD) { mkey[idx] = INT_MIN; den[idx] = 0.0f; }
    if (idx < n_nodes * HID) agg[idx] = 0.0f;
}

// Pass 1: scores + running segment max (via int-encoded atomicMax).
__global__ void edge_score_kernel(const float* __restrict__ q, int qld, int qofs,
                                  const float* __restrict__ khat,
                                  const int* __restrict__ esrc,
                                  const int* __restrict__ edst,
                                  const float* __restrict__ prel, // [H]
                                  float* __restrict__ sc,
                                  int* __restrict__ mkey, int nE)
{
    int idx = blockIdx.x * blockDim.x + threadIdx.x;
    if (idx >= nE * NHEAD) return;
    int e = idx >> 3, h = idx & 7;
    int s = esrc[e], d = edst[e];
    const float* qp = q + (size_t)d * qld + qofs + h * DHEAD;
    const float* kp = khat + (size_t)s * HID + h * DHEAD;
    float acc = 0.0f;
#pragma unroll
    for (int i = 0; i < DHEAD; ++i) acc += qp[i] * kp[i];
    acc *= prel[h] * 0.25f;   // 1/sqrt(16)
    sc[idx] = acc;
    atomicMax(&mkey[d * NHEAD + h], fkey_encode(acc));
}

// Pass 2: e = exp(sc - m[dst]); denominator via atomicAdd.
__global__ void edge_exp_kernel(float* __restrict__ sc,
                                const int* __restrict__ edst,
                                const int* __restrict__ mkey,
                                float* __restrict__ den, int nE)
{
    int idx = blockIdx.x * blockDim.x + threadIdx.x;
    if (idx >= nE * NHEAD) return;
    int e = idx >> 3, h = idx & 7;
    int d = edst[e];
    float m  = fkey_decode(mkey[d * NHEAD + h]);
    float ev = expf(sc[idx] - m);
    sc[idx] = ev;
    atomicAdd(&den[d * NHEAD + h], ev);
}

// Pass 3: agg[dst] += vhat[src] * (e / den[dst])
__global__ void edge_agg_kernel(const float* __restrict__ sc,
                                const float* __restrict__ vhat,
                                const int* __restrict__ esrc,
                                const int* __restrict__ edst,
                                const float* __restrict__ den,
                                float* __restrict__ agg, int nE)
{
    int idx = blockIdx.x * blockDim.x + threadIdx.x;
    if (idx >= nE * NHEAD) return;
    int e = idx >> 3, h = idx & 7;
    int s = esrc[e], d = edst[e];
    float alpha = sc[idx] / (den[d * NHEAD + h] + 1e-16f);
    const float* vp = vhat + (size_t)s * HID + h * DHEAD;
    float* op       = agg  + (size_t)d * HID + h * DHEAD;
#pragma unroll
    for (int f = 0; f < DHEAD; ++f) atomicAdd(op + f, vp[f] * alpha);
}

__global__ void gelu_kernel(const float* __restrict__ in,
                            float* __restrict__ out, int n)
{
    int idx = blockIdx.x * blockDim.x + threadIdx.x;
    if (idx >= n) return;
    float x = in[idx];
    out[idx] = 0.5f * x * (1.0f + erff(x * 0.70710678118654752f)); // exact gelu
}

// ---------------------------------------------------------------------------
extern "C" void kernel_launch(void* const* d_in, const int* in_sizes, int n_in,
                              void* d_out, int out_size, void* d_ws, size_t ws_size,
                              hipStream_t stream)
{
    const float* x_user     = (const float*)d_in[0];
    const float* x_item     = (const float*)d_in[1];
    const int*   esrc_ui    = (const int*)d_in[2];
    const int*   edst_ui    = (const int*)d_in[3];
    const int*   esrc_iu    = (const int*)d_in[4];
    const int*   edst_iu    = (const int*)d_in[5];
    const int*   esrc_uu    = (const int*)d_in[6];
    const int*   edst_uu    = (const int*)d_in[7];
    const float* W_in_user  = (const float*)d_in[8];
    const float* b_in_user  = (const float*)d_in[9];
    const float* W_in_item  = (const float*)d_in[10];
    const float* b_in_item  = (const float*)d_in[11];
    const float* W_kqv_user = (const float*)d_in[12];
    const float* b_kqv_user = (const float*)d_in[13];
    const float* W_kqv_item = (const float*)d_in[14];
    const float* b_kqv_item = (const float*)d_in[15];
    const float* W_out_user = (const float*)d_in[16];
    const float* b_out_user = (const float*)d_in[17];
    const float* W_out_item = (const float*)d_in[18];
    const float* b_out_item = (const float*)d_in[19];
    const float* skip_user  = (const float*)d_in[20];
    const float* skip_item  = (const float*)d_in[21];
    const float* A_k        = (const float*)d_in[22];
    const float* A_v        = (const float*)d_in[23];
    const float* p_rel      = (const float*)d_in[24];
    const float* W_lin      = (const float*)d_in[25];
    const float* b_lin      = (const float*)d_in[26];

    const int NU   = in_sizes[0] / 128;
    const int NI   = in_sizes[1] / 64;
    const int E_ui = in_sizes[2];
    const int E_iu = in_sizes[4];
    const int E_uu = in_sizes[6];
    const int L    = 2;

    // ---- workspace layout (floats) ----
    float* w = (float*)d_ws;
    size_t off = 0;
    auto alloc = [&](size_t n) { float* p = w + off; off += n; return p; };
    float* xu      = alloc((size_t)NU * HID);
    float* xi      = alloc((size_t)NI * HID);
    float* kqv_u   = alloc((size_t)NU * 384);
    float* kqv_i   = alloc((size_t)NI * 384);
    float* khat_ui = alloc((size_t)NU * HID);
    float* vhat_ui = alloc((size_t)NU * HID);
    float* khat_uu = alloc((size_t)NU * HID);
    float* vhat_uu = alloc((size_t)NU * HID);
    float* khat_iu = alloc((size_t)NI * HID);
    float* vhat_iu = alloc((size_t)NI * HID);
    float* sc_ui   = alloc((size_t)E_ui * NHEAD);
    float* sc_iu   = alloc((size_t)E_iu * NHEAD);
    float* sc_uu   = alloc((size_t)E_uu * NHEAD);
    int*   mkey_u  = (int*)alloc((size_t)NU * NHEAD);
    float* den_u   = alloc((size_t)NU * NHEAD);
    int*   mkey_i  = (int*)alloc((size_t)NI * NHEAD);
    float* den_i   = alloc((size_t)NI * NHEAD);
    float* agg_u   = alloc((size_t)NU * HID);
    float* agg_i   = alloc((size_t)NI * HID);
    // gelu buffers alias khat (dead once edge passes of the layer are done)
    float* gbuf_u = khat_ui;
    float* gbuf_i = khat_iu;

    dim3 blk(256);
    auto gemm = [&](const float* A, int lda, const float* B, int ldb,
                    const float* bias, float* C, int ldc,
                    const float* Xprev, const float* skip,
                    int M, int N, int K, int mode) {
        dim3 grid(N / BN, (M + BM - 1) / BM);
        gemm_bf16_wmma<<<grid, blk, 0, stream>>>(A, lda, B, ldb, bias, C, ldc,
                                                 Xprev, skip, M, N, K, mode);
    };

    // input projections + relu
    gemm(x_user, 128, W_in_user, HID, b_in_user, xu, HID, nullptr, nullptr, NU, HID, 128, 1);
    gemm(x_item, 64,  W_in_item, HID, b_in_item, xi, HID, nullptr, nullptr, NI, HID, 64, 1);

    const size_t rsz = (size_t)NHEAD * DHEAD * DHEAD; // 2048 per (l, r)

    for (int l = 0; l < L; ++l) {
        // KQV projections
        gemm(xu, HID, W_kqv_user + (size_t)l * HID * 384, 384, b_kqv_user + l * 384,
             kqv_u, 384, nullptr, nullptr, NU, 384, HID, 0);
        gemm(xi, HID, W_kqv_item + (size_t)l * HID * 384, 384, b_kqv_item + l * 384,
             kqv_i, 384, nullptr, nullptr, NI, 384, HID, 0);

        // per-node relation transforms (hoisted per-edge einsum)
        auto xf = [&](const float* src, int colofs, const float* Am, float* out, int n) {
            int tot = n * HID;
            head_transform_kernel<<<dim3((tot + 255) / 256), blk, 0, stream>>>(
                src, 384, colofs, Am, out, n);
        };
        xf(kqv_u, 0,   A_k + (size_t)(l * 3 + 0) * rsz, khat_ui, NU);
        xf(kqv_u, 256, A_v + (size_t)(l * 3 + 0) * rsz, vhat_ui, NU);
        xf(kqv_u, 0,   A_k + (size_t)(l * 3 + 2) * rsz, khat_uu, NU);
        xf(kqv_u, 256, A_v + (size_t)(l * 3 + 2) * rsz, vhat_uu, NU);
        xf(kqv_i, 0,   A_k + (size_t)(l * 3 + 1) * rsz, khat_iu, NI);
        xf(kqv_i, 256, A_v + (size_t)(l * 3 + 1) * rsz, vhat_iu, NI);

        init_agg_kernel<<<dim3((NU * HID + 255) / 256), blk, 0, stream>>>(mkey_u, den_u, agg_u, NU);
        init_agg_kernel<<<dim3((NI * HID + 255) / 256), blk, 0, stream>>>(mkey_i, den_i, agg_i, NI);

        // pass 1: scores + segment max
        auto score = [&](const float* q, const float* khat, const int* es, const int* ed,
                         const float* ph, float* sc, int* mk, int nE) {
            edge_score_kernel<<<dim3((nE * NHEAD + 255) / 256), blk, 0, stream>>>(
                q, 384, 128, khat, es, ed, ph, sc, mk, nE);
        };
        score(kqv_i, khat_ui, esrc_ui, edst_ui, p_rel + (l * 3 + 0) * NHEAD, sc_ui, mkey_i, E_ui);
        score(kqv_u, khat_iu, esrc_iu, edst_iu, p_rel + (l * 3 + 1) * NHEAD, sc_iu, mkey_u, E_iu);
        score(kqv_u, khat_uu, esrc_uu, edst_uu, p_rel + (l * 3 + 2) * NHEAD, sc_uu, mkey_u, E_uu);

        // pass 2: exp + denominator
        auto eexp = [&](float* sc, const int* ed, const int* mk, float* den, int nE) {
            edge_exp_kernel<<<dim3((nE * NHEAD + 255) / 256), blk, 0, stream>>>(sc, ed, mk, den, nE);
        };
        eexp(sc_ui, edst_ui, mkey_i, den_i, E_ui);
        eexp(sc_iu, edst_iu, mkey_u, den_u, E_iu);
        eexp(sc_uu, edst_uu, mkey_u, den_u, E_uu);

        // pass 3: weighted scatter-add of vhat
        auto eagg = [&](const float* sc, const float* vh, const int* es, const int* ed,
                        const float* den, float* agg, int nE) {
            edge_agg_kernel<<<dim3((nE * NHEAD + 255) / 256), blk, 0, stream>>>(
                sc, vh, es, ed, den, agg, nE);
        };
        eagg(sc_ui, vhat_ui, esrc_ui, edst_ui, den_i, agg_i, E_ui);
        eagg(sc_iu, vhat_iu, esrc_iu, edst_iu, den_u, agg_u, E_iu);
        eagg(sc_uu, vhat_uu, esrc_uu, edst_uu, den_u, agg_u, E_uu);

        // gelu -> output projection -> sigmoid-gated skip -> relu (in-place on xu/xi)
        gelu_kernel<<<dim3((NU * HID + 255) / 256), blk, 0, stream>>>(agg_u, gbuf_u, NU * HID);
        gelu_kernel<<<dim3((NI * HID + 255) / 256), blk, 0, stream>>>(agg_i, gbuf_i, NI * HID);
        gemm(gbuf_u, HID, W_out_user + (size_t)l * HID * HID, HID, b_out_user + l * HID,
             xu, HID, xu, skip_user + l, NU, HID, HID, 2);
        gemm(gbuf_i, HID, W_out_item + (size_t)l * HID * HID, HID, b_out_item + l * HID,
             xi, HID, xi, skip_item + l, NI, HID, HID, 2);
    }

    // final shared linear; users then items into d_out [NU+NI, 64]
    float* outf = (float*)d_out;
    gemm(xu, HID, W_lin, 64, b_lin, outf, 64, nullptr, nullptr, NU, 64, HID, 0);
    gemm(xi, HID, W_lin, 64, b_lin, outf + (size_t)NU * 64, 64, nullptr, nullptr, NI, 64, HID, 0);
}
